// LSDAN_9122510537179
// MI455X (gfx1250) — matrix operand order repaired and checked by reading, not compiled, not run on hardware
//
#include <hip/hip_runtime.h>
#include <hip/hip_bf16.h>

typedef __attribute__((ext_vector_type(2)))  float        v2f;
typedef __attribute__((ext_vector_type(8)))  float        v8f;
typedef __attribute__((ext_vector_type(16))) __bf16       v16bf;
typedef __attribute__((ext_vector_type(8)))  __bf16       v8bf;
typedef __attribute__((ext_vector_type(4)))  unsigned int v4u;
typedef __attribute__((ext_vector_type(8)))  int          v8i;
typedef __attribute__((ext_vector_type(4)))  int          v4i;

#define NN     4096
#define IN_F   256
#define HH     128
#define ALPHA  0.2f
#define MASKV  -1000.0f

// ---------------------------------------------------------------------------
// Tensor Data Mover: 2-D tile load (bf16 elements) global -> LDS.
// D# layout per CDNA5 ISA ch.8 (group0 128b, group1 256b; groups2/3 zero = 2D).
// Toolchain uses the 6-arg builtin form (extra int32x8 group before cpol).
// ---------------------------------------------------------------------------
__device__ __forceinline__ void tdm_load_bf16_2d(
    const void* gptr,        // global byte address of tile start
    unsigned    lds_off,     // LDS byte offset of tile destination
    unsigned    tile_w,      // tile dim0 (contiguous elements)
    unsigned    tile_h,      // tile dim1 (rows)
    unsigned    tensor_w,    // tensor dim0
    unsigned    tensor_h,    // tensor dim1
    unsigned long long row_stride)  // dim0 stride in elements
{
    unsigned long long ga = (unsigned long long)gptr;
    v4u g0;
    g0[0] = 1u;                                    // count=1, no gather/restore
    g0[1] = lds_off;                               // lds_addr
    g0[2] = (unsigned)(ga & 0xFFFFFFFFu);          // global_addr[31:0]
    g0[3] = (unsigned)((ga >> 32) & 0x1FFFFFFu)    // global_addr[56:32]
          | (2u << 30);                            // type=2 ("image")
    v8i g1;
    g1[0] = (int)(1u << 16);                       // data_size=1 (2 bytes), mask=0
    g1[1] = (int)((tensor_w & 0xFFFFu) << 16);     // tensor_dim0[15:0]
    g1[2] = (int)((tensor_w >> 16) | ((tensor_h & 0xFFFFu) << 16));
    g1[3] = (int)((tensor_h >> 16) | (tile_w << 16));      // tile_dim0
    g1[4] = (int)(tile_h & 0xFFFFu);               // tile_dim1; tile_dim2=0
    g1[5] = (int)(unsigned)(row_stride & 0xFFFFFFFFull);   // dim0_stride[31:0]
    g1[6] = (int)(unsigned)((row_stride >> 32) & 0xFFFFull);
    g1[7] = 0;
    v4i z4 = {0, 0, 0, 0};
    v8i z8 = {0, 0, 0, 0, 0, 0, 0, 0};
    __builtin_amdgcn_tensor_load_to_lds(g0, g1, z4, z4, z8, 0);
}

// ---------------------------------------------------------------------------
// FP32 GEMM via V_WMMA_F32_16X16X4_F32.
// C[M x N] = A[M x K] * B  (B is [K x N] row-major, or [N x K] if TRANSB).
// One wave per block; wave computes a 16 x (16*NT) tile of C.
// ---------------------------------------------------------------------------
template<int NT, bool TRANSB, bool ACCUM>
__global__ __launch_bounds__(32)
void gemm_f32_wmma(const float* __restrict__ A, const float* __restrict__ B,
                   float* __restrict__ C, int K, int lda, int ldb, int ldc)
{
    const int lane = threadIdx.x;
    const int l16  = lane & 15;
    const int half = lane >> 4;
    const int m0   = blockIdx.y * 16;
    const int n0   = blockIdx.x * (16 * NT);

    v8f acc[NT] = {};
    if (ACCUM) {
        #pragma unroll
        for (int t = 0; t < NT; ++t)
            #pragma unroll
            for (int r = 0; r < 8; ++r)
                acc[t][r] = C[(size_t)(m0 + r + 8 * half) * ldc + n0 + t * 16 + l16];
    }

    const float* Arow = A + (size_t)(m0 + l16) * lda;
    for (int k = 0; k < K; k += 4) {
        __builtin_prefetch(Arow + k + 128, 0, 1);
        v2f a;
        a.x = Arow[k + 2 * half];
        a.y = Arow[k + 2 * half + 1];
        #pragma unroll
        for (int t = 0; t < NT; ++t) {
            const int ncol = n0 + t * 16 + l16;
            v2f b;
            if (TRANSB) {
                const float* Bp = B + (size_t)ncol * ldb + k + 2 * half;
                b.x = Bp[0];
                b.y = Bp[1];
            } else {
                b.x = B[(size_t)(k + 2 * half)     * ldb + ncol];
                b.y = B[(size_t)(k + 2 * half + 1) * ldb + ncol];
            }
            acc[t] = __builtin_amdgcn_wmma_f32_16x16x4_f32(
                false, a, false, b, (short)0, acc[t], false, false);
        }
    }

    #pragma unroll
    for (int t = 0; t < NT; ++t)
        #pragma unroll
        for (int r = 0; r < 8; ++r)
            C[(size_t)(m0 + r + 8 * half) * ldc + n0 + t * 16 + l16] = acc[t][r];
}

// ---------------------------------------------------------------------------
// BF16 GEMM, TDM-fed and LDS double-buffered, for adjacency powers A^2, A^3.
// 256 threads = 8 waves; workgroup computes 64x64 of C; wave computes 16x32.
// Per K-stage (32), the Tensor Data Mover DMAs a 64x32 A tile and a 32x64
// B tile into LDS while the waves run WMMAs on the previous stage.
// ---------------------------------------------------------------------------
__global__ __launch_bounds__(256)
void gemm_bf16_tdm(const __bf16* __restrict__ A, const __bf16* __restrict__ B,
                   __bf16* __restrict__ C, int K, int lda, int ldb, int ldc)
{
    __shared__ __align__(16) __bf16 As[2][64][32];   // [stage][row][k]
    __shared__ __align__(16) __bf16 Bs[2][32][64];   // [stage][k][col]

    const int tid  = threadIdx.x;
    const int wave = tid >> 5;
    const int lane = tid & 31;
    const int l16  = lane & 15;
    const int half = lane >> 4;

    const int m0 = blockIdx.y * 64;
    const int n0 = blockIdx.x * 64;
    const int mw = (wave >> 1) * 16;          // wave's row offset inside tile
    const int nw = (wave & 1) * 32;           // wave's col offset inside tile

    const unsigned ldsA0 = (unsigned)(uintptr_t)&As[0][0][0];
    const unsigned ldsA1 = (unsigned)(uintptr_t)&As[1][0][0];
    const unsigned ldsB0 = (unsigned)(uintptr_t)&Bs[0][0][0];
    const unsigned ldsB1 = (unsigned)(uintptr_t)&Bs[1][0][0];

    const int S = K / 32;

    // Prologue: stage 0 DMA (wave 0, single issue; TDM ignores EXEC).
    if (tid == 0) {
        tdm_load_bf16_2d(A + (size_t)m0 * lda, ldsA0, 32u, 64u,
                         (unsigned)lda, (unsigned)NN, (unsigned long long)lda);
        tdm_load_bf16_2d(B + n0,              ldsB0, 64u, 32u,
                         (unsigned)ldb, (unsigned)NN, (unsigned long long)ldb);
    }

    v8f acc0 = {}, acc1 = {};

    for (int s = 0; s < S; ++s) {
        const int buf = s & 1;

        // Issue next stage into the other buffer (safe: that buffer was last
        // read in stage s-1 and every wave passed the end-of-stage barrier).
        if (tid == 0 && (s + 1) < S) {
            const int k1 = (s + 1) * 32;
            tdm_load_bf16_2d(A + (size_t)m0 * lda + k1,
                             buf ? ldsA0 : ldsA1, 32u, 64u,
                             (unsigned)lda, (unsigned)NN, (unsigned long long)lda);
            tdm_load_bf16_2d(B + (size_t)k1 * ldb + n0,
                             buf ? ldsB0 : ldsB1, 64u, 32u,
                             (unsigned)ldb, (unsigned)NN, (unsigned long long)ldb);
        }
        // Wave 0 waits for stage-s pair; pairs complete in order, so with the
        // next pair outstanding TENSORcnt<=2 guarantees stage s has landed.
        if (tid < 32) {
            if ((s + 1) < S) __builtin_amdgcn_s_wait_tensorcnt(2);
            else             __builtin_amdgcn_s_wait_tensorcnt(0);
        }
        __syncthreads();

        // A fragment (16x32 bf16): two contiguous 16-byte LDS reads per lane.
        // af elements 0..7  = A[m, 16*half + 0..7]
        // af elements 8..15 = A[m, 16*half + 8 + 0..7]
        union { v16bf v; v8bf q[2]; } af;
        af.q[0] = *(const v8bf*)&As[buf][mw + l16][16 * half];
        af.q[1] = *(const v8bf*)&As[buf][mw + l16][16 * half + 8];

        // B fragments (32x16 bf16): bf.h[2v(+1)] = B[2v(+1) + 16*half][n]
        union { v16bf v; __bf16 h[16]; } bf0, bf1;
        #pragma unroll
        for (int v = 0; v < 8; ++v) {
            const int kr = 2 * v + 16 * half;
            bf0.h[2 * v]     = Bs[buf][kr]    [nw + l16];
            bf0.h[2 * v + 1] = Bs[buf][kr + 1][nw + l16];
            bf1.h[2 * v]     = Bs[buf][kr]    [nw + 16 + l16];
            bf1.h[2 * v + 1] = Bs[buf][kr + 1][nw + 16 + l16];
        }

        acc0 = __builtin_amdgcn_wmma_f32_16x16x32_bf16(
            false, af.v, false, bf0.v, (short)0, acc0, false, false);
        acc1 = __builtin_amdgcn_wmma_f32_16x16x32_bf16(
            false, af.v, false, bf1.v, (short)0, acc1, false, false);

        __syncthreads();
    }

    #pragma unroll
    for (int r = 0; r < 8; ++r) {
        const size_t row = (size_t)(m0 + mw + r + 8 * half) * ldc;
        C[row + n0 + nw + l16]      = (__bf16)acc0[r];
        C[row + n0 + nw + 16 + l16] = (__bf16)acc1[r];
    }
}

// ---------------------------------------------------------------------------
__global__ void f32_to_bf16(const float* __restrict__ in, __bf16* __restrict__ out, int n)
{
    int i = blockIdx.x * blockDim.x + threadIdx.x;
    if (i < n) out[i] = (__bf16)in[i];
}

__global__ void zero_f32(float* __restrict__ p, int n)
{
    int i = blockIdx.x * blockDim.x + threadIdx.x;
    if (i < n) p[i] = 0.0f;
}

// s1[i] = W1h[i,:] . r[0:H] ; s2[i] = W1h[i,:] . r[H:2H]
__global__ void s_vec(const float* __restrict__ W1h, const float* __restrict__ r,
                      float* __restrict__ s1, float* __restrict__ s2)
{
    int i = blockIdx.x * blockDim.x + threadIdx.x;
    if (i >= NN) return;
    const float* row = W1h + (size_t)i * HH;
    float a1 = 0.f, a2 = 0.f;
    #pragma unroll 8
    for (int t = 0; t < HH; ++t) {
        float w = row[t];
        a1 += w * r[t];
        a2 += w * r[HH + t];
    }
    s1[i] = a1;
    s2[i] = a2;
}

// attention row: softmax_j( mask ? -1000 : leakyrelu(s1[i]+s2[j]) )
__global__ __launch_bounds__(256)
void attn_softmax(const __bf16* __restrict__ Ak, const float* __restrict__ s1,
                  const float* __restrict__ s2, float* __restrict__ P)
{
    __shared__ float red[256];
    const int row = blockIdx.x, tid = threadIdx.x;
    const __bf16* arow = Ak + (size_t)row * NN;
    float* prow = P + (size_t)row * NN;
    const float s1v = s1[row];

    float mx = -3.0e38f;
    for (int j = tid; j < NN; j += 256) {
        float e = s1v + s2[j];
        e = e > 0.f ? e : ALPHA * e;
        float v = ((float)arow[j] != 0.f) ? e : MASKV;
        mx = fmaxf(mx, v);
    }
    red[tid] = mx; __syncthreads();
    for (int s = 128; s > 0; s >>= 1) {
        if (tid < s) red[tid] = fmaxf(red[tid], red[tid + s]);
        __syncthreads();
    }
    mx = red[0]; __syncthreads();

    float sum = 0.f;
    for (int j = tid; j < NN; j += 256) {
        float e = s1v + s2[j];
        e = e > 0.f ? e : ALPHA * e;
        float v = ((float)arow[j] != 0.f) ? e : MASKV;
        float p = __expf(v - mx);
        prow[j] = p;
        sum += p;
    }
    red[tid] = sum; __syncthreads();
    for (int s = 128; s > 0; s >>= 1) {
        if (tid < s) red[tid] += red[tid + s];
        __syncthreads();
    }
    const float rinv = 1.f / red[0];
    for (int j = tid; j < NN; j += 256) prow[j] *= rinv;
}

// in-place: S[row,:] = softmax_j( mask ? -1000 : S[row,j] )
__global__ __launch_bounds__(256)
void score_softmax(const __bf16* __restrict__ Ak, float* __restrict__ S)
{
    __shared__ float red[256];
    const int row = blockIdx.x, tid = threadIdx.x;
    const __bf16* arow = Ak + (size_t)row * NN;
    float* srow = S + (size_t)row * NN;

    float mx = -3.0e38f;
    for (int j = tid; j < NN; j += 256) {
        float v = ((float)arow[j] != 0.f) ? srow[j] : MASKV;
        mx = fmaxf(mx, v);
    }
    red[tid] = mx; __syncthreads();
    for (int s = 128; s > 0; s >>= 1) {
        if (tid < s) red[tid] = fmaxf(red[tid], red[tid + s]);
        __syncthreads();
    }
    mx = red[0]; __syncthreads();

    float sum = 0.f;
    for (int j = tid; j < NN; j += 256) {
        float v = ((float)arow[j] != 0.f) ? srow[j] : MASKV;
        float p = __expf(v - mx);
        srow[j] = p;
        sum += p;
    }
    red[tid] = sum; __syncthreads();
    for (int s = 128; s > 0; s >>= 1) {
        if (tid < s) red[tid] += red[tid + s];
        __syncthreads();
    }
    const float rinv = 1.f / red[0];
    for (int j = tid; j < NN; j += 256) srow[j] *= rinv;
}

// d_out = concat(U + O, O)
__global__ void final_out(const float* __restrict__ U, const float* __restrict__ O,
                          float* __restrict__ out, int n)
{
    int i = blockIdx.x * blockDim.x + threadIdx.x;
    if (i < n) {
        float o = O[i];
        out[i]     = U[i] + o;
        out[n + i] = o;
    }
}

// ---------------------------------------------------------------------------
extern "C" void kernel_launch(void* const* d_in, const int* in_sizes, int n_in,
                              void* d_out, int out_size, void* d_ws, size_t ws_size,
                              hipStream_t stream)
{
    const float* X   = (const float*)d_in[0];   // [4096 x 256]
    const float* Afp = (const float*)d_in[1];   // [4096 x 4096]
    const float* U   = (const float*)d_in[2];   // [4096 x 128]
    const float* W1  = (const float*)d_in[3];   // [128 x 256]
    const float* W2  = (const float*)d_in[4];   // [128 x 256]
    const float* r   = (const float*)d_in[5];   // [256]

    char* w = (char*)d_ws;
    float*  W1h    = (float*) (w + ((size_t)0   << 20));   // 2 MB
    float*  W2h    = (float*) (w + ((size_t)2   << 20));   // 2 MB
    float*  hk     = (float*) (w + ((size_t)4   << 20));   // 2 MB
    float*  outAcc = (float*) (w + ((size_t)6   << 20));   // 2 MB
    float*  s1p    = (float*) (w + ((size_t)8   << 20));   // 16 KB
    float*  s2p    = (float*) (w + ((size_t)8   << 20) + (1 << 16));
    __bf16* Abf    = (__bf16*)(w + ((size_t)16  << 20));   // 32 MB
    __bf16* A2bf   = (__bf16*)(w + ((size_t)48  << 20));   // 32 MB
    __bf16* A3bf   = (__bf16*)(w + ((size_t)80  << 20));   // 32 MB
    float*  P      = (float*) (w + ((size_t)112 << 20));   // 64 MB

    const int NH = NN * HH;            // 524288
    const int NA = NN * NN;            // 16777216

    // Projections: W1h = X @ W1^T, W2h = X @ W2^T  (M=4096, K=256, N=128)
    gemm_f32_wmma<8, true, false><<<dim3(1, NN / 16), 32, 0, stream>>>(
        X, W1, W1h, IN_F, IN_F, IN_F, HH);
    gemm_f32_wmma<8, true, false><<<dim3(1, NN / 16), 32, 0, stream>>>(
        X, W2, W2h, IN_F, IN_F, IN_F, HH);

    // s1/s2 from W1h and r
    s_vec<<<NN / 256, 256, 0, stream>>>(W1h, r, s1p, s2p);

    // adjacency -> bf16 (0/1, exact)
    f32_to_bf16<<<NA / 256, 256, 0, stream>>>(Afp, Abf, NA);

    // output accumulator must start at zero every call
    zero_f32<<<NH / 256, 256, 0, stream>>>(outAcc, NH);

    const __bf16* Ak[3] = { Abf, A2bf, A3bf };

    for (int hop = 0; hop < 3; ++hop) {
        if (hop == 1) {
            // A2 = A @ A   (bf16 in, f32 accum, bf16 out; TDM-fed tiles)
            gemm_bf16_tdm<<<dim3(NN / 64, NN / 64), 256, 0, stream>>>(
                Abf, Abf, A2bf, NN, NN, NN, NN);
        } else if (hop == 2) {
            // A3 = A2 @ A
            gemm_bf16_tdm<<<dim3(NN / 64, NN / 64), 256, 0, stream>>>(
                A2bf, Abf, A3bf, NN, NN, NN, NN);
        }
        const __bf16* Akb = Ak[hop];

        // attention = softmax(mask ? -1000 : leakyrelu(s1+s2))   -> P
        attn_softmax<<<NN, 256, 0, stream>>>(Akb, s1p, s2p, P);

        // h_k = attention @ W1h    (M=4096, K=4096, N=128)
        gemm_f32_wmma<8, false, false><<<dim3(1, NN / 16), 32, 0, stream>>>(
            P, W1h, hk, NN, NN, HH, HH);

        // scores = h_k @ W2h^T     (M=4096, K=128, N=4096) -> reuse P
        gemm_f32_wmma<8, true, false><<<dim3(NN / 128, NN / 16), 32, 0, stream>>>(
            hk, W2h, P, HH, HH, HH, NN);

        // attn_w = softmax(mask ? -1000 : scores) in place
        score_softmax<<<NN, 256, 0, stream>>>(Akb, P);

        // output += attn_w @ h_k   (M=4096, K=4096, N=128), accumulate
        gemm_f32_wmma<8, false, true><<<dim3(1, NN / 16), 32, 0, stream>>>(
            P, hk, outAcc, NN, NN, HH, HH);
    }

    final_out<<<NH / 256, 256, 0, stream>>>(U, outAcc, (float*)d_out, NH);
}
